// DIN_4234837754264
// MI455X (gfx1250) — matrix operand order repaired and checked by reading, not compiled, not run on hardware
//
#include <hip/hip_runtime.h>
#include <hip/hip_bf16.h>
#include <math.h>

// ---------------------------------------------------------------------------
// DIN (Deep Interest Network) inference for MI455X / gfx1250.
//
// Kernel 1 (din_attention_pool): one wave32 per (branch, batch-item).
//   - 16 seq rows per tile, 13 tiles cover L=200 (tail rows masked).
//   - Attention layer 1 (64 -> 16) via two chained v_wmma_f32_16x16x32_f16
//     (K=64 split into two K=32 steps), bias preloaded into the accumulator.
//   - Dice activation in-register on the WMMA accumulator (fast v_rcp sigmoid).
//   - Layers 2+3 (16->8->1) via wave-private LDS transpose + VALU, with the
//     8 output neurons split across lane halves and recombined via shfl_xor.
//   - Wave-private LDS ordering enforced with s_wait_dscnt (no block barrier
//     in the hot loop -> waves drift freely and hide HBM latency).
//   - Pooling fused: each seq element is read from HBM exactly once.
// Kernel 2 (din_final_mlp): one thread per batch row, 80->32->16->1 head.
// ---------------------------------------------------------------------------

typedef __attribute__((ext_vector_type(16))) _Float16 v16h;
typedef __attribute__((ext_vector_type(8)))  float    v8f;

#define B_     4096
#define L_     200
#define D_     16
#define WPB    8          // waves per block (256 threads)
#define NTILE  13         // ceil(200 / 16)

__device__ __forceinline__ float fast_sigmoid(float x) {
    // v_exp_f32 + v_rcp_f32 instead of the IEEE div ladder.
    return __builtin_amdgcn_rcpf(1.0f + __expf(-x));
}

__device__ __forceinline__ float dice_f(float x, float alpha) {
    // inference-mode Dice: p = sigmoid(x); y = p*x + (1-p)*alpha*x
    const float p = fast_sigmoid(x);
    return x * (p + (1.0f - p) * alpha);
}

// Wave-local LDS fence: LDS ops of one wave are processed in order (DScnt),
// so draining DScnt makes all lanes' prior stores visible to the wave's own
// subsequent loads. The "memory" clobber stops compiler reordering.
__device__ __forceinline__ void wave_lds_sync() {
    asm volatile("s_wait_dscnt 0x0" ::: "memory");
}

__global__ __launch_bounds__(256, 4)
void din_attention_pool_kernel(
    const float* __restrict__ seq1, const float* __restrict__ tgt1,
    const float* __restrict__ seq2, const float* __restrict__ tgt2,
    const int*   __restrict__ len1, const int*   __restrict__ len2,
    const float* __restrict__ aW1,  const float* __restrict__ ab1,
    const float* __restrict__ aal1,
    const float* __restrict__ aW2,  const float* __restrict__ ab2,
    const float* __restrict__ aal2,
    const float* __restrict__ aW3,  const float* __restrict__ ab3,
    float* __restrict__ pool_out)   // [2][B][16] in d_ws
{
    __shared__ float h1s[WPB][16][17];   // +1 pad: conflict-free transpose
    __shared__ float w2s[16][8];
    __shared__ float b2s[8], al2s[8], w3s[8];
    __shared__ float b3s;

    const int tid = threadIdx.x;
    if (tid < 128)                     w2s[tid >> 3][tid & 7] = aW2[tid];
    if (tid >= 128 && tid < 136) {
        const int j = tid - 128;
        b2s[j] = ab2[j]; al2s[j] = aal2[j]; w3s[j] = aW3[j];
    }
    if (tid == 136) b3s = ab3[0];
    __syncthreads();                     // weights visible to all 8 waves

    const int wv     = tid >> 5;
    const int lane   = tid & 31;
    const int task   = blockIdx.x * WPB + wv;       // 0 .. 2*B-1
    const int branch = task >> 12;                  // task / 4096
    const int b      = task & (B_ - 1);

    const float* __restrict__ seq = branch ? seq2 : seq1;
    const float* __restrict__ tgt = branch ? tgt2 : tgt1;
    const int    len              = branch ? len2[b] : len1[b];

    const int n  = lane & 15;   // WMMA C-layout column; also row id for L2/L3
    const int hi = lane >> 4;   // lane half
    const int d0 = hi * 8;      // which 8 of the 16 dims this lane carries

    // Target-embedding half for this lane (loop invariant).
    float t8[8];
    #pragma unroll
    for (int k = 0; k < 8; ++k) t8[k] = tgt[b * D_ + d0 + k];

    // Layer-1 weights in WMMA B-layout (32x16 f16):
    //   lanes 0-15 hold K = 0..7,16..23 of column n, lanes 16-31 hold K = 8..15,24..31.
    // bw0 covers feature rows 0..31 ([t, s]), bw1 rows 32..63 ([t-s, t*s]).
    v16h bw0, bw1;
    const int kr = hi * 8;
    #pragma unroll
    for (int e = 0; e < 16; ++e) {
        bw0[e] = (_Float16)aW1[(kr + e + (e >= 8 ? 8 : 0))      * 16 + n];
        bw1[e] = (_Float16)aW1[(kr + e + (e >= 8 ? 8 : 0) + 32) * 16 + n];
    }

    const float bias1  = ab1[n];
    const float alpha1 = aal1[n];
    const int   j0     = hi * 4;         // this half's layer-2 neuron range

    float poolAcc[8];
    #pragma unroll
    for (int k = 0; k < 8; ++k) poolAcc[k] = 0.0f;

    const size_t rowbase = (size_t)b * L_ * D_;

    for (int tIdx = 0; tIdx < NTILE; ++tIdx) {
        const int l  = tIdx * 16 + n;
        const int lc = (l < L_) ? l : (L_ - 1);      // clamp tail loads

        // One HBM touch per seq element: 2 x b128 per lane.
        const float* sp = seq + rowbase + (size_t)lc * D_ + d0;
        const float4 sA = *reinterpret_cast<const float4*>(sp);
        const float4 sB = *reinterpret_cast<const float4*>(sp + 4);
        float s8[8] = {sA.x, sA.y, sA.z, sA.w, sB.x, sB.y, sB.z, sB.w};

        // Prefetch next tile (emits global_prefetch_b8).
        if (tIdx + 1 < NTILE) {
            int ln = l + 16; ln = (ln < L_) ? ln : (L_ - 1);
            __builtin_prefetch(seq + rowbase + (size_t)ln * D_ + d0, 0, 3);
        }

        // A-matrix fragments (16x32 f16 each). Lane layout makes the four
        // feature groups line up with [t | s] and [t-s | t*s] halves.
        v16h a0, a1;
        #pragma unroll
        for (int k = 0; k < 8; ++k) {
            a0[k]     = (_Float16)t8[k];
            a0[k + 8] = (_Float16)s8[k];
            a1[k]     = (_Float16)(t8[k] - s8[k]);
            a1[k + 8] = (_Float16)(t8[k] * s8[k]);
        }

        // Layer 1: C = bias; C += A0*B0; C += A1*B1.
        v8f c;
        #pragma unroll
        for (int i = 0; i < 8; ++i) c[i] = bias1;
        c = __builtin_amdgcn_wmma_f32_16x16x32_f16(false, a0, false, bw0,
                                                   (short)0, c, false, false);
        c = __builtin_amdgcn_wmma_f32_16x16x32_f16(false, a1, false, bw1,
                                                   (short)0, c, false, false);

        // Dice on accumulator (column bias/alpha == per-lane), stage to LDS.
        #pragma unroll
        for (int i = 0; i < 8; ++i) {
            const int m = i + hi * 8;               // C layout: VGPRi -> row
            h1s[wv][m][n] = dice_f(c[i], alpha1);
        }
        wave_lds_sync();                            // wave-private RAW fence

        // Layers 2+3 per seq-row m == n; each lane half evaluates 4 of the
        // 8 hidden neurons, halves are recombined with one shfl_xor.
        float h1row[16];
        #pragma unroll
        for (int k = 0; k < 16; ++k) h1row[k] = h1s[wv][n][k];

        float partial = 0.0f;
        #pragma unroll
        for (int jj = 0; jj < 4; ++jj) {
            const int j = j0 + jj;
            float acc = b2s[j];
            #pragma unroll
            for (int k = 0; k < 16; ++k) acc += h1row[k] * w2s[k][j];
            partial += dice_f(acc, al2s[j]) * w3s[j];
        }
        float score = partial + __shfl_xor(partial, 16, 32) + b3s;
        if (l >= len || l >= L_) score = 0.0f;       // length + tail mask

        // Fused pooling: this lane owns seq[l][d0..d0+7] already.
        #pragma unroll
        for (int k = 0; k < 8; ++k) poolAcc[k] += score * s8[k];
        wave_lds_sync();                             // h1s reused next tile
    }

    // Cross-row reduction of the pooled partials through wave-private LDS.
    #pragma unroll
    for (int k = 0; k < 8; ++k) h1s[wv][n][d0 + k] = poolAcc[k];
    wave_lds_sync();
    if (lane < 16) {
        float acc = 0.0f;
        #pragma unroll
        for (int m = 0; m < 16; ++m) acc += h1s[wv][m][lane];
        pool_out[((size_t)branch * B_ + b) * D_ + lane] = acc;
    }
}

__global__ __launch_bounds__(256)
void din_final_mlp_kernel(
    const float* __restrict__ user, const float* __restrict__ tgt1,
    const float* __restrict__ tgt2, const float* __restrict__ pool, // [2][B][16]
    const float* __restrict__ mW1, const float* __restrict__ mb1,
    const float* __restrict__ mal1,
    const float* __restrict__ mW2, const float* __restrict__ mb2,
    const float* __restrict__ mal2,
    const float* __restrict__ mW3, const float* __restrict__ mb3,
    float* __restrict__ out)
{
    const int b = blockIdx.x * blockDim.x + threadIdx.x;
    if (b >= B_) return;

    float emb[80];
    #pragma unroll
    for (int k = 0; k < 16; ++k) {
        emb[k]      = user[b * 16 + k];
        emb[16 + k] = pool[b * 16 + k];                       // pool1
        emb[32 + k] = tgt1[b * 16 + k];
        emb[48 + k] = pool[(size_t)B_ * 16 + b * 16 + k];     // pool2
        emb[64 + k] = tgt2[b * 16 + k];
    }

    float h1[32];
    for (int j = 0; j < 32; ++j) {
        float acc = mb1[j];
        for (int k = 0; k < 80; ++k) acc += emb[k] * mW1[k * 32 + j];
        h1[j] = dice_f(acc, mal1[j]);
    }
    float h2[16];
    for (int j = 0; j < 16; ++j) {
        float acc = mb2[j];
        for (int k = 0; k < 32; ++k) acc += h1[k] * mW2[k * 16 + j];
        h2[j] = dice_f(acc, mal2[j]);
    }
    float acc = mb3[0];
    for (int k = 0; k < 16; ++k) acc += h2[k] * mW3[k];
    out[b] = fast_sigmoid(acc);
}

extern "C" void kernel_launch(void* const* d_in, const int* in_sizes, int n_in,
                              void* d_out, int out_size, void* d_ws, size_t ws_size,
                              hipStream_t stream) {
    const float* user = (const float*)d_in[0];
    const float* seq1 = (const float*)d_in[1];
    const float* tgt1 = (const float*)d_in[2];
    const float* seq2 = (const float*)d_in[3];
    const float* tgt2 = (const float*)d_in[4];
    const int*   len1 = (const int*)  d_in[5];
    const int*   len2 = (const int*)  d_in[6];
    const float* aW1  = (const float*)d_in[7];
    const float* ab1  = (const float*)d_in[8];
    const float* aal1 = (const float*)d_in[9];
    const float* aW2  = (const float*)d_in[10];
    const float* ab2  = (const float*)d_in[11];
    const float* aal2 = (const float*)d_in[12];
    const float* aW3  = (const float*)d_in[13];
    const float* ab3  = (const float*)d_in[14];
    const float* mW1  = (const float*)d_in[15];
    const float* mb1  = (const float*)d_in[16];
    const float* mal1 = (const float*)d_in[17];
    const float* mW2  = (const float*)d_in[18];
    const float* mb2  = (const float*)d_in[19];
    const float* mal2 = (const float*)d_in[20];
    const float* mW3  = (const float*)d_in[21];
    const float* mb3  = (const float*)d_in[22];

    float* pool = (float*)d_ws;        // [2][B][16] f32 = 512 KB scratch
    float* out  = (float*)d_out;       // [B] f32

    // 2*B wave-tasks / 8 waves per block = 1024 blocks.
    din_attention_pool_kernel<<<(2 * B_) / WPB, 256, 0, stream>>>(
        seq1, tgt1, seq2, tgt2, len1, len2,
        aW1, ab1, aal1, aW2, ab2, aal2, aW3, ab3, pool);

    din_final_mlp_kernel<<<(B_ + 255) / 256, 256, 0, stream>>>(
        user, tgt1, tgt2, pool,
        mW1, mb1, mal1, mW2, mb2, mal2, mW3, mb3, out);
}
